// FlexAttention_18923625906354
// MI455X (gfx1250) — compile-verified, matrix operand order checked by reading, not run on hardware
//
#include <hip/hip_runtime.h>
#include <math.h>

typedef __bf16 bf16;
typedef __attribute__((ext_vector_type(16))) __bf16 v16bf;
typedef __attribute__((ext_vector_type(8)))  __bf16 v8bf;
typedef __attribute__((ext_vector_type(8)))  float  v8f;
typedef __attribute__((ext_vector_type(4)))  unsigned u32x4;
typedef __attribute__((ext_vector_type(8)))  unsigned u32x8;

#define HIDDEN 1024
#define SEQ    1024
#define BATCH  4
#define NQH    32
#define NKVH   16
#define HD     64
#define QDIM   2048
#define KVDIM  1024
#define TOKENS (BATCH*SEQ)       /* 4096 */
#define QKV_N  (QDIM + 2*KVDIM)  /* 4096 */

// LDS tile geometry for the TDM-staged GEMM:
// 128 rows x 64 bf16 per tile; TDM pads 16B after each 128B row -> stride 72.
#define TILE_R   128
#define TILE_K   64
#define LDS_STR  72            /* elements per row in LDS (64 + 8 pad) */
#define TILE_ELT (TILE_R * LDS_STR)

// ---------- helpers ----------
__device__ __forceinline__ bf16 f2bf(float f){
  unsigned u = __builtin_bit_cast(unsigned, f);
  unsigned r = u + 0x7FFFu + ((u >> 16) & 1u);   // round-to-nearest-even
  unsigned short s = (unsigned short)(r >> 16);
  return __builtin_bit_cast(bf16, s);
}

// Build a 16-element bf16 fragment from two contiguous 16-byte chunks.
// Matches the CDNA5 16-bit A/B operand swizzle: per lane, halves 0..7 come
// from one 8-element chunk, halves 8..15 from another.
__device__ __forceinline__ v16bf ldfrag(const bf16* lo, const bf16* hi){
  v8bf a = *(const v8bf*)lo;
  v8bf b = *(const v8bf*)hi;
  return __builtin_shufflevector(a, b, 0,1,2,3,4,5,6,7,8,9,10,11,12,13,14,15);
}

__device__ __forceinline__ float rmax16(float x){
  #pragma unroll
  for (int m = 1; m < 16; m <<= 1) x = fmaxf(x, __shfl_xor(x, m, 16));
  return x;
}
__device__ __forceinline__ float rsum16(float x){
  #pragma unroll
  for (int m = 1; m < 16; m <<= 1) x += __shfl_xor(x, m, 16);
  return x;
}

// ---------- Tensor Data Mover: 2-D tile (Global -> LDS), bf16 elements ----------
// D# per CDNA5 ISA. 2 SGPR groups (<=2-D tensor).
// pad_enable inserts 4 dwords (16B) after every 32 dwords (one 128B tile row),
// giving an LDS row stride of 144B = 72 bf16 to break bank conflicts.
__device__ __forceinline__ void tdm_load_2d(unsigned lds_off_bytes, const bf16* gptr,
                                            unsigned tens_d0, unsigned tens_d1,
                                            unsigned stride_d0)
{
  unsigned long long ga = (unsigned long long)(size_t)gptr;
  u32x4 g0;
  g0[0] = 1u;                                            // count=1 (valid), user mode
  g0[1] = lds_off_bytes;                                 // lds_addr
  g0[2] = (unsigned)(ga & 0xFFFFFFFFu);                  // global_addr[31:0]
  g0[3] = (unsigned)((ga >> 32) & 0x01FFFFFFu)           // global_addr[56:32]
        | (2u << 30);                                    // type=2 ("image")
  u32x8 g1;
  g1[0] = (1u << 16)                                     // data_size=1 -> 2 bytes
        | (1u << 20)                                     // pad_enable
        | (4u << 22)                                     // pad_interval: 32 dwords
        | (3u << 25);                                    // pad_amount: 4 dwords
  g1[1] = (tens_d0 & 0xFFFFu) << 16;                     // tensor_dim0[15:0]
  g1[2] = (tens_d0 >> 16) | ((tens_d1 & 0xFFFFu) << 16); // dim0[31:16] | dim1[15:0]
  g1[3] = (tens_d1 >> 16) | ((unsigned)TILE_K << 16);    // dim1[31:16] | tile_dim0
  g1[4] = (unsigned)TILE_R;                              // tile_dim1 (tile_dim2=0)
  g1[5] = stride_d0;                                     // tensor_dim0_stride[31:0]
  g1[6] = 0u;                                            // stride[47:32] | dim1_stride lo
  g1[7] = 0u;
  asm volatile("tensor_load_to_lds %0, %1" :: "s"(g0), "s"(g1) : "memory");
}

// ---------- conversion kernels ----------
__global__ void k_conv_in(const float* __restrict__ in, bf16* __restrict__ out, int n){
  int i = blockIdx.x * 256 + threadIdx.x;
  if (i < n) out[i] = f2bf(in[i]);
}

// in[rows][cols] (f32) -> out[cols][rows] (bf16): stores W^T so the GEMM B
// operand is N x K row-major (same per-lane swizzle as the A operand).
__global__ void k_transpose_bf16(const float* __restrict__ in, bf16* __restrict__ out,
                                 int rows, int cols){
  int i = blockIdx.x * 256 + threadIdx.x;
  if (i < rows * cols){
    int r = i / cols, c = i % cols;
    out[(size_t)c * rows + r] = f2bf(in[i]);
  }
}

// ---------- bf16 WMMA GEMM with TDM-staged LDS tiles ----------
// C[M,N] = A[M,K] * Bt[N,K]^T. Block = 8 waves, block tile 128x128, K-step 64.
// Wave 0 drives the Tensor Data Mover (double-buffered A/B tiles); all waves
// consume fragments from LDS. EPI=0: f32 C. EPI=1: QKV split epilogue.
template<int EPI>
__global__ void __launch_bounds__(256)
k_gemm(const bf16* __restrict__ A, const bf16* __restrict__ Bt,
       int M, int N, int K,
       float* __restrict__ Cf,
       bf16* __restrict__ Qb, bf16* __restrict__ Kb, bf16* __restrict__ Vt)
{
  // [buf 0/1][A=0,B=1][128][72]
  __shared__ __align__(16) bf16 smem[2 * 2 * TILE_ELT];

  const int lane = threadIdx.x & 31;
  const int w    = threadIdx.x >> 5;     // 8 waves
  const int half = lane >> 4;
  const int lr   = lane & 15;
  const int wm   = w & 3;                // 4 waves along M (32 rows each)
  const int wn   = w >> 2;               // 2 waves along N (64 cols each)
  const int rowB = blockIdx.y * 128;     // block row
  const int colB = blockIdx.x * 128;     // block col
  const int steps = K / TILE_K;

  v8f acc[2][4];
  #pragma unroll
  for (int i = 0; i < 2; i++)
    #pragma unroll
    for (int j = 0; j < 4; j++) acc[i][j] = {};

  // prologue: stage tile pair 0 into buffer 0
  if (w == 0){
    tdm_load_2d(0u * TILE_ELT * 2u,            A  + (size_t)rowB * K, (unsigned)K, (unsigned)M, (unsigned)K);
    tdm_load_2d(1u * TILE_ELT * 2u,            Bt + (size_t)colB * K, (unsigned)K, (unsigned)N, (unsigned)K);
  }

  int cur = 0;
  for (int it = 0; it < steps; it++){
    if (w == 0){
      if (it + 1 < steps){
        const unsigned nb = (unsigned)(cur ^ 1);
        tdm_load_2d((nb * 2u + 0u) * TILE_ELT * 2u, A  + (size_t)rowB * K + (it + 1) * TILE_K,
                    (unsigned)K, (unsigned)M, (unsigned)K);
        tdm_load_2d((nb * 2u + 1u) * TILE_ELT * 2u, Bt + (size_t)colB * K + (it + 1) * TILE_K,
                    (unsigned)K, (unsigned)N, (unsigned)K);
        __builtin_amdgcn_s_wait_tensorcnt(2);   // pair 'it' fully in LDS
      } else {
        __builtin_amdgcn_s_wait_tensorcnt(0);
      }
    }
    __syncthreads();                            // tile 'it' visible to all waves

    const bf16* As = &smem[(cur * 2 + 0) * TILE_ELT];
    const bf16* Bs = &smem[(cur * 2 + 1) * TILE_ELT];

    #pragma unroll
    for (int kk = 0; kk < TILE_K; kk += 32){
      v16bf af[2], bfr[4];
      #pragma unroll
      for (int i = 0; i < 2; i++){
        const bf16* p = As + (size_t)(wm * 32 + i * 16 + lr) * LDS_STR + kk;
        af[i] = ldfrag(p + half * 8, p + 16 + half * 8);
      }
      #pragma unroll
      for (int j = 0; j < 4; j++){
        const bf16* p = Bs + (size_t)(wn * 64 + j * 16 + lr) * LDS_STR + kk;
        bfr[j] = ldfrag(p + half * 8, p + 16 + half * 8);
      }
      #pragma unroll
      for (int i = 0; i < 2; i++)
        #pragma unroll
        for (int j = 0; j < 4; j++)
          acc[i][j] = __builtin_amdgcn_wmma_f32_16x16x32_bf16(
              false, af[i], false, bfr[j], (short)0, acc[i][j], false, false);
    }

    __syncthreads();                            // everyone done with buf 'cur'
    cur ^= 1;
  }

  // C layout: vgpr v, lane -> row = v + 8*(lane>>4), col = lane&15
  const int row0 = rowB + wm * 32;
  const int col0 = colB + wn * 64;
  #pragma unroll
  for (int i = 0; i < 2; i++){
    #pragma unroll
    for (int j = 0; j < 4; j++){
      const int n = col0 + j * 16 + lr;
      #pragma unroll
      for (int v = 0; v < 8; v++){
        const int m = row0 + i * 16 + v + 8 * half;
        float val = acc[i][j][v];
        if (EPI == 0){
          Cf[(size_t)m * N + n] = val;
        } else {
          const int b = m >> 10, s = m & 1023;
          bf16 bv = f2bf(val);
          if (n < QDIM){                       // Q: [b][h][s][d]
            int hh = n >> 6, d = n & 63;
            Qb[(((size_t)b * NQH + hh) * SEQ + s) * HD + d] = bv;
          } else if (n < QDIM + KVDIM){        // K: [b][kvh][s][d]
            int kvh = (n - QDIM) >> 6, d = n & 63;
            Kb[(((size_t)b * NKVH + kvh) * SEQ + s) * HD + d] = bv;
          } else {                             // V transposed: [b][kvh][d][s]
            int kvh = (n - QDIM - KVDIM) >> 6, d = n & 63;
            Vt[(((size_t)b * NKVH + kvh) * HD + d) * SEQ + s] = bv;
          }
        }
      }
    }
  }
}

// ---------- flash attention: 1 wave per (b, h, 16-row q tile) ----------
__global__ void __launch_bounds__(32)
k_attn(const bf16* __restrict__ Qb, const bf16* __restrict__ Kb,
       const bf16* __restrict__ Vt, bf16* __restrict__ X)
{
  __shared__ __align__(16) bf16 pl[16 * 32];   // P tile staging (C-layout -> A-layout)

  const int lane = threadIdx.x;
  const int half = lane >> 4, lr = lane & 15;
  const int qt  = blockIdx.x;                  // 0..63
  const int h   = blockIdx.y;                  // 0..31
  const int b   = blockIdx.z;                  // 0..3
  const int kvh = h >> 1;                      // GQA: 2 q-heads per kv-head
  const int qb  = qt * 16;
  const float slope = exp2f(-0.25f * (float)(h + 1));   // ALiBi, 32 heads

  // Q A-fragments: 16x64 -> two 16x32 frags
  const bf16* Qp = Qb + (((size_t)b * NQH + h) * SEQ + qb) * HD;
  v16bf qf[2];
  #pragma unroll
  for (int f = 0; f < 2; f++){
    const bf16* p = Qp + (size_t)lr * HD + f * 32;
    qf[f] = ldfrag(p + half * 8, p + 16 + half * 8);
  }

  v8f o[4];
  #pragma unroll
  for (int t = 0; t < 4; t++) o[t] = {};
  float mrow[8], lrow[8];
  #pragma unroll
  for (int v = 0; v < 8; v++){ mrow[v] = -1e30f; lrow[v] = 0.f; }

  const bf16* Kp0 = Kb + ((size_t)b * NKVH + kvh) * SEQ * HD;   // [s][d]
  const bf16* Vp0 = Vt + ((size_t)b * NKVH + kvh) * HD * SEQ;   // [d][s]

  for (int jb = 0; jb <= qb + 15; jb += 32){
    __builtin_prefetch(Kp0 + (size_t)(jb + 32 + lr) * HD, 0, 0);

    // ---- S = Q * K^T over 32 keys (two 16x16 f32 tiles) ----
    v8f s0 = {}, s1 = {};
    {
      const bf16* kp  = Kp0 + (size_t)(jb + lr) * HD;
      const bf16* kp2 = Kp0 + (size_t)(jb + 16 + lr) * HD;
      v16bf k00 = ldfrag(kp  + half * 8,       kp  + 16 + half * 8);
      v16bf k01 = ldfrag(kp  + 32 + half * 8,  kp  + 48 + half * 8);
      v16bf k10 = ldfrag(kp2 + half * 8,       kp2 + 16 + half * 8);
      v16bf k11 = ldfrag(kp2 + 32 + half * 8,  kp2 + 48 + half * 8);
      s0 = __builtin_amdgcn_wmma_f32_16x16x32_bf16(false, qf[0], false, k00, (short)0, s0, false, false);
      s0 = __builtin_amdgcn_wmma_f32_16x16x32_bf16(false, qf[1], false, k01, (short)0, s0, false, false);
      s1 = __builtin_amdgcn_wmma_f32_16x16x32_bf16(false, qf[0], false, k10, (short)0, s1, false, false);
      s1 = __builtin_amdgcn_wmma_f32_16x16x32_bf16(false, qf[1], false, k11, (short)0, s1, false, false);
    }

    // ---- scale + ALiBi bias + causal mask ----
    float sa[8], sb[8];
    #pragma unroll
    for (int v = 0; v < 8; v++){
      const int qrow = qb + v + 8 * half;
      const int key0 = jb + lr;
      const int key1 = jb + 16 + lr;
      float x0 = s0[v] * 0.125f + slope * (float)(key0 - qrow);
      float x1 = s1[v] * 0.125f + slope * (float)(key1 - qrow);
      sa[v] = (key0 > qrow) ? -1e30f : x0;
      sb[v] = (key1 > qrow) ? -1e30f : x1;
    }

    // ---- online softmax (row = v + 8*half spans a 16-lane group) ----
    float alpha[8];
    #pragma unroll
    for (int v = 0; v < 8; v++){
      float mx = rmax16(fmaxf(sa[v], sb[v]));
      float mnew = fmaxf(mrow[v], mx);
      alpha[v] = __expf(mrow[v] - mnew);
      mrow[v] = mnew;
      float p0 = __expf(sa[v] - mnew);
      float p1 = __expf(sb[v] - mnew);
      sa[v] = p0; sb[v] = p1;
      lrow[v] = lrow[v] * alpha[v] + rsum16(p0 + p1);
    }
    #pragma unroll
    for (int t = 0; t < 4; t++)
      #pragma unroll
      for (int v = 0; v < 8; v++) o[t][v] *= alpha[v];

    // ---- P (C-layout) -> LDS -> A-fragment layout ----
    #pragma unroll
    for (int v = 0; v < 8; v++){
      const int row = v + 8 * half;
      pl[row * 32 + lr]      = f2bf(sa[v]);
      pl[row * 32 + 16 + lr] = f2bf(sb[v]);
    }
    __syncthreads();
    v16bf pa;
    {
      const bf16* p = pl + lr * 32;
      pa = ldfrag(p + half * 8, p + 16 + half * 8);
    }
    __syncthreads();

    // ---- O += P * V (contract 32 keys; V^T rows contiguous in key) ----
    #pragma unroll
    for (int t = 0; t < 4; t++){
      const bf16* vp = Vp0 + (size_t)(t * 16 + lr) * SEQ + jb;
      v16bf vb = ldfrag(vp + half * 8, vp + 16 + half * 8);
      o[t] = __builtin_amdgcn_wmma_f32_16x16x32_bf16(false, pa, false, vb, (short)0, o[t], false, false);
    }
  }

  // ---- normalize and store X[b][s][h][d] (bf16) ----
  #pragma unroll
  for (int t = 0; t < 4; t++){
    #pragma unroll
    for (int v = 0; v < 8; v++){
      const int srow = qb + v + 8 * half;
      const int d = t * 16 + lr;
      float val = o[t][v] / lrow[v];
      X[(((size_t)b * SEQ + srow) * NQH + h) * HD + d] = f2bf(val);
    }
  }
}

// ---------- launcher ----------
extern "C" void kernel_launch(void* const* d_in, const int* in_sizes, int n_in,
                              void* d_out, int out_size, void* d_ws, size_t ws_size,
                              hipStream_t stream)
{
  (void)in_sizes; (void)n_in; (void)out_size; (void)ws_size;
  const float* inputs = (const float*)d_in[0];   // [4,1024,1024]
  const float* Wqkv   = (const float*)d_in[1];   // [1024,4096]
  const float* Wout   = (const float*)d_in[2];   // [2048,1024]
  float* out = (float*)d_out;

  char* ws = (char*)d_ws;
  bf16* Ain   = (bf16*)(ws);                        //  8 MiB: input bf16 [4096][1024]
  bf16* WqkvT = (bf16*)(ws + (size_t)( 8u << 20));  //  8 MiB: [4096][1024]
  bf16* WoutT = (bf16*)(ws + (size_t)(16u << 20));  //  4 MiB: [1024][2048]
  bf16* Qb    = (bf16*)(ws + (size_t)(20u << 20));  // 16 MiB: [4][32][1024][64]
  bf16* Kb    = (bf16*)(ws + (size_t)(36u << 20));  //  8 MiB: [4][16][1024][64]
  bf16* Vt    = (bf16*)(ws + (size_t)(44u << 20));  //  8 MiB: [4][16][64][1024]
  bf16* Xb    = (bf16*)(ws + (size_t)(52u << 20));  // 16 MiB: [4096][2048]

  k_conv_in<<<(TOKENS * HIDDEN + 255) / 256, 256, 0, stream>>>(inputs, Ain, TOKENS * HIDDEN);
  k_transpose_bf16<<<(HIDDEN * QKV_N + 255) / 256, 256, 0, stream>>>(Wqkv, WqkvT, HIDDEN, QKV_N);
  k_transpose_bf16<<<(QDIM * HIDDEN + 255) / 256, 256, 0, stream>>>(Wout, WoutT, QDIM, HIDDEN);

  // GEMM1: [4096,1024] x [1024,4096] with QKV-split epilogue (TDM-staged)
  dim3 g1(QKV_N / 128, TOKENS / 128);
  k_gemm<1><<<g1, 256, 0, stream>>>(Ain, WqkvT, TOKENS, QKV_N, HIDDEN,
                                    nullptr, Qb, Kb, Vt);

  // Flash attention: one wave per (q-tile, head, batch)
  dim3 ga(SEQ / 16, NQH, BATCH);
  k_attn<<<ga, 32, 0, stream>>>(Qb, Kb, Vt, Xb);

  // GEMM2: [4096,2048] x [2048,1024] -> f32 output (TDM-staged)
  dim3 g2(HIDDEN / 128, TOKENS / 128);
  k_gemm<0><<<g2, 256, 0, stream>>>(Xb, WoutT, TOKENS, HIDDEN, QDIM,
                                    out, nullptr, nullptr, nullptr);
}
